// CLIPAttention_44581760533134
// MI455X (gfx1250) — matrix operand order repaired and checked by reading, not compiled, not run on hardware
//
#include <hip/hip_runtime.h>
#include <hip/hip_bf16.h>

// CLIP attention for MI455X (gfx1250), wave32 + WMMA f16 (f32 accumulate),
// with CDNA5 async global->LDS tile staging (ASYNCcnt) and double buffering.
// B=8, S=1024, E=1024, H=16, D=64.

typedef _Float16 v8h  __attribute__((ext_vector_type(8)));
typedef _Float16 v16h __attribute__((ext_vector_type(16)));
typedef float    v8f  __attribute__((ext_vector_type(8)));

#define EMBED 1024
#define SEQ   1024
#define BATCH 8
#define HEADS 16
#define HDIM  64

__device__ __forceinline__ v8f wmma_f16(v16h a, v16h b, v8f c) {
    // D = A(16x32 f16) * B(32x16 f16) + C(16x16 f32)
    return __builtin_amdgcn_wmma_f32_16x16x32_f16(
        /*neg_a=*/false, a, /*neg_b=*/false, b,
        /*c_mod=*/(short)0, c, /*reuse_a=*/false, /*reuse_b=*/false);
}

// Async DMA: 16 bytes per lane, global -> LDS, tracked by ASYNCcnt.
// lptr is a generic pointer into the (shared) destination tile; its low 32
// bits are the LDS byte address (ISA 10.2: LDS aperture truncation). Passing
// the full pointer as an extra asm operand makes the shared object escape so
// the "memory" clobber keeps the subsequent ds reads live.
__device__ __forceinline__ void async_load_b128(_Float16* lptr, const void* gptr) {
    unsigned lds_off = (unsigned)(unsigned long long)lptr;
    asm volatile("global_load_async_to_lds_b128 %0, %1, off"
                 :: "v"(lds_off), "v"(gptr), "v"(lptr) : "memory");
}
__device__ __forceinline__ void wait_asynccnt0() {
    asm volatile("s_wait_asynccnt 0" ::: "memory");
}

// A-fragment loader, 16-bit A 16x32 layout (ISA 7.12.2):
// lanes 0-15 hold K{0..7, 16..23}, lanes 16-31 hold K{8..15, 24..31}.
// p points at rowBase + k0 (contraction-contiguous); kb = (lane>>4)*8.
__device__ __forceinline__ v16h load_a16(const _Float16* p, int kb) {
    v8h lo = *(const v8h*)(p + kb);
    v8h hi = *(const v8h*)(p + kb + 16);
    return __builtin_shufflevector(lo, hi, 0,1,2,3,4,5,6,7,8,9,10,11,12,13,14,15);
}

__global__ void f32_to_f16_kernel(const float* __restrict__ in,
                                  _Float16* __restrict__ out, int n) {
    int i = blockIdx.x * blockDim.x + threadIdx.x;
    if (i < n) out[i] = (_Float16)in[i];
}

// C = A[MxK] * W[NxK]^T + bias (torch Linear). Block = 8 waves; each wave a
// 16x64 tile (same n-range): the 64x32 W tile is staged once per k-step into
// LDS via async DMA and shared by all 8 waves (8x traffic reduction).
// mode 0: f16 row-major out.  mode 1: f16 V head-transposed out
// Vt[((b*H+h)*64+d)*S + s].   mode 2: f32 row-major out (final).
__global__ void gemm_wmma_kernel(const _Float16* __restrict__ A,
                                 const _Float16* __restrict__ W,
                                 const float* __restrict__ bias,
                                 _Float16* __restrict__ outH,
                                 float* __restrict__ outF,
                                 int M, int N, int K, int mode) {
    const int lane = threadIdx.x & 31;
    const int wave = threadIdx.x >> 5;
    const int hi   = lane >> 4;      // lane-half select
    const int ln   = lane & 15;
    const int m0   = blockIdx.x * 128 + wave * 16;
    const int n0   = blockIdx.y * 64;
    const int kb   = hi * 8;

    __shared__ __align__(16) _Float16 smW[2][64 * 32];   // 2 x 4KB W tiles

    // Cooperative tile fetch: 256 threads x 16B = 4KB = 64 rows x 32 halves.
    const int wr  = threadIdx.x >> 2;   // W row within tile (0..63)
    const int wsg = threadIdx.x & 3;    // 8-half segment   (0..3)
    const _Float16* wsrc = W + (size_t)(n0 + wr) * K + wsg * 8;
    auto issue_w = [&](int k0, int buf) {
        async_load_b128(&smW[buf][wr * 32 + wsg * 8], wsrc + k0);
    };

    v8f acc[4] = {};
    const _Float16* arow = A + (size_t)(m0 + ln) * K;

    issue_w(0, 0);
    v16h af = load_a16(arow, kb);              // software-pipelined A fragment
    for (int k0 = 0, it = 0; k0 < K; k0 += 32, ++it) {
        const int cur = it & 1;
        wait_asynccnt0();         // my DMA into smW[cur] landed
        __syncthreads();          // everyone's landed; prev buffer free
        if (k0 + 32 < K) issue_w(k0 + 32, cur ^ 1);

        __builtin_prefetch(arow + k0 + 128, 0, 3);   // stream A ahead
        // Preload all B fragments (8x ds_load_b128 in flight), then the
        // 4-WMMA chain drains them with partial dscnt waits.
        const _Float16* sw = &smW[cur][0];
        v16h bf[4];
#pragma unroll
        for (int t = 0; t < 4; ++t)
            bf[t] = load_a16(sw + (t * 16 + ln) * 32, kb);

        v16h afn = (k0 + 32 < K) ? load_a16(arow + k0 + 32, kb) : af;
#pragma unroll
        for (int t = 0; t < 4; ++t)
            acc[t] = wmma_f16(af, bf[t], acc[t]);
        af = afn;
    }

#pragma unroll
    for (int t = 0; t < 4; ++t) {
#pragma unroll
        for (int r = 0; r < 8; ++r) {
            int m = m0 + r + 8 * hi;          // C/D layout: M = r + 8*(lane>=16)
            int n = n0 + t * 16 + ln;         //             N = lane%16
            float v = acc[t][r] + bias[n];
            if (mode == 2) {
                outF[(size_t)m * N + n] = v;
            } else if (mode == 1) {
                int b = m >> 10, s = m & 1023;
                int h = n >> 6,  d = n & 63;
                outH[(((size_t)(b * HEADS + h) * HDIM + d) << 10) + s] = (_Float16)v;
            } else {
                outH[(size_t)m * N + n] = (_Float16)v;
            }
        }
    }
}

// Flash attention. Block = 8 waves over 128 queries of one (b,h); all waves
// sweep the same keys, so the 32-key K tile (32x64 f16) and V tile (64x32
// d-major f16) are async-staged into LDS once per step and shared.
// Computes S^T = K_tile @ Q^T so post-softmax P lands directly in the WMMA
// A-fragment layout (no transpose needed).
__global__ void attn_wmma_kernel(const _Float16* __restrict__ Qh,
                                 const _Float16* __restrict__ Kh,
                                 const _Float16* __restrict__ Vt,
                                 _Float16* __restrict__ Oh) {
    const int lane = threadIdx.x & 31;
    const int wave = threadIdx.x >> 5;
    const int hi   = lane >> 4;
    const int ln   = lane & 15;
    const int bh   = blockIdx.y;            // b*16 + h
    const int b    = bh >> 4, h = bh & 15;
    const int q0   = blockIdx.x * 128 + wave * 16;
    const int kb8  = hi * 8;
    const float scale = 0.125f;             // 1/sqrt(64)

    __shared__ __align__(16) _Float16 smem[2][4096];  // [K tile 2048 | V tile 2048]

    const _Float16* kbase = Kh + (size_t)(b * SEQ) * EMBED + h * HDIM;
    const _Float16* vbase = Vt + (size_t)bh * HDIM * SEQ;

    // Cooperative fetch: K tile = 32 tokens x 64 halves (token-major);
    //                    V tile = 64 d-rows x 32 key-halves (d-major).
    const int ktok = threadIdx.x >> 3, kseg = threadIdx.x & 7;   // 32 x 8seg
    const int vrow = threadIdx.x >> 2, vseg = threadIdx.x & 3;   // 64 x 4seg
    const _Float16* ksrc = kbase + (size_t)ktok * EMBED + kseg * 8;
    const _Float16* vsrc = vbase + (size_t)vrow * SEQ + vseg * 8;
    auto issue_kv = [&](int k0, int buf) {
        async_load_b128(&smem[buf][ktok * 64 + kseg * 8], ksrc + (size_t)k0 * EMBED);
        async_load_b128(&smem[buf][2048 + vrow * 32 + vseg * 8], vsrc + k0);
    };

    // Q as B-fragment (B[d][q] = Q[q][d]): lane = query col, table-order d.
    const _Float16* qrow = Qh + ((size_t)(b * SEQ + q0 + ln)) * EMBED + h * HDIM;
    v16h qb0 = load_a16(qrow, kb8);        // d chunk 0..31
    v16h qb1 = load_a16(qrow + 32, kb8);   // d chunk 32..63

    float m_q = -3.0e38f;   // running max for query ln
    float l_q = 0.0f;       // running denom for query ln
    v8f acc[4] = {};        // 16 q x 64 d output tile

    issue_kv(0, 0);
    for (int k0 = 0, it = 0; k0 < SEQ; k0 += 32, ++it) {
        const int cur = it & 1;
        wait_asynccnt0();
        __syncthreads();
        if (k0 + 32 < SEQ) issue_kv(k0 + 32, cur ^ 1);

        const _Float16* sk = &smem[cur][0];     // [tok][d]
        const _Float16* sv = &smem[cur][2048];  // [d][key]

        // Preload all K A-fragments, then the score WMMA chain.
        const _Float16* kr0 = sk + (size_t)ln * 64;
        const _Float16* kr1 = sk + (size_t)(16 + ln) * 64;
        v16h ka0 = load_a16(kr0, kb8);
        v16h ka1 = load_a16(kr0 + 32, kb8);
        v16h ka2 = load_a16(kr1, kb8);
        v16h ka3 = load_a16(kr1 + 32, kb8);

        v8f s0 = {}, s1 = {};
        s0 = wmma_f16(ka0, qb0, s0);
        s0 = wmma_f16(ka1, qb1, s0);
        s1 = wmma_f16(ka2, qb0, s1);
        s1 = wmma_f16(ka3, qb1, s1);

        // Issue the V B-fragment ds loads now: their latency is hidden under
        // the softmax VALU/transcendental stream below.
        v16h vf[4];
#pragma unroll
        for (int dt = 0; dt < 4; ++dt)
            vf[dt] = load_a16(sv + (dt * 16 + ln) * 32, kb8);

        // Online softmax for query ln: this lane holds 8 keys per tile; its
        // partner lane (lane ^ 16) holds the other 8.
        float t0[8], t1[8];
        float mx = -3.0e38f;
#pragma unroll
        for (int r = 0; r < 8; ++r) {
            t0[r] = s0[r] * scale;
            t1[r] = s1[r] * scale;
            mx = fmaxf(mx, fmaxf(t0[r], t1[r]));
        }
        mx = fmaxf(mx, __shfl_xor(mx, 16, 32));
        float m_new = fmaxf(m_q, mx);
        float fac = __expf(m_q - m_new);
        m_q = m_new;

        // P in A-fragment layout: e0..7 = this lane's s0 keys, e8..15 = s1.
        v16h P;
        float rsum = 0.0f;
#pragma unroll
        for (int r = 0; r < 8; ++r) {
            float p0 = __expf(t0[r] - m_new);
            float p1 = __expf(t1[r] - m_new);
            rsum += p0 + p1;
            P[r]     = (_Float16)p0;
            P[r + 8] = (_Float16)p1;
        }
        rsum += __shfl_xor(rsum, 16, 32);
        l_q = l_q * fac + rsum;

        // Rescale accumulator; acc reg r belongs to query r+8*hi, whose
        // factor lives in lane r+8*hi.
#pragma unroll
        for (int r = 0; r < 8; ++r) {
            float fr = __shfl(fac, r + 8 * hi, 32);
            acc[0][r] *= fr; acc[1][r] *= fr; acc[2][r] *= fr; acc[3][r] *= fr;
        }

        // P(16x32) @ V(32x64).
#pragma unroll
        for (int dt = 0; dt < 4; ++dt)
            acc[dt] = wmma_f16(P, vf[dt], acc[dt]);
    }

    // Normalize and store f16 [B*S, E] row-major (A input of output proj).
#pragma unroll
    for (int r = 0; r < 8; ++r) {
        float lr  = __shfl(l_q, r + 8 * hi, 32);
        float inv = 1.0f / lr;
        int m = q0 + r + 8 * hi;
        _Float16* orow = Oh + ((size_t)(b * SEQ + m)) * EMBED + h * HDIM;
#pragma unroll
        for (int dt = 0; dt < 4; ++dt)
            orow[dt * 16 + ln] = (_Float16)(acc[dt][r] * inv);
    }
}

extern "C" void kernel_launch(void* const* d_in, const int* in_sizes, int n_in,
                              void* d_out, int out_size, void* d_ws, size_t ws_size,
                              hipStream_t stream) {
    const float* x  = (const float*)d_in[0];
    const float* wq = (const float*)d_in[1];
    const float* bq = (const float*)d_in[2];
    const float* wk = (const float*)d_in[3];
    const float* bk = (const float*)d_in[4];
    const float* wv = (const float*)d_in[5];
    const float* bv = (const float*)d_in[6];
    const float* wo = (const float*)d_in[7];
    const float* bo = (const float*)d_in[8];
    float* out = (float*)d_out;

    const size_t MB = 1024u * 1024u;
    char* ws = (char*)d_ws;
    _Float16* xh  = (_Float16*)(ws + 0 * MB);   // 16 MB  x in f16
    _Float16* wqh = (_Float16*)(ws + 16 * MB);  //  2 MB
    _Float16* wkh = (_Float16*)(ws + 18 * MB);
    _Float16* wvh = (_Float16*)(ws + 20 * MB);
    _Float16* woh = (_Float16*)(ws + 22 * MB);
    _Float16* Qh  = (_Float16*)(ws + 24 * MB);  // 16 MB
    _Float16* Kh  = (_Float16*)(ws + 40 * MB);  // 16 MB
    _Float16* Vt  = (_Float16*)(ws + 56 * MB);  // 16 MB, head-transposed
    _Float16* Ah  = (_Float16*)(ws + 72 * MB);  // 16 MB, attn output

    const int NX = BATCH * SEQ * EMBED;   // 8388608
    const int NW = EMBED * EMBED;         // 1048576

    f32_to_f16_kernel<<<(NX + 255) / 256, 256, 0, stream>>>(x,  xh,  NX);
    f32_to_f16_kernel<<<(NW + 255) / 256, 256, 0, stream>>>(wq, wqh, NW);
    f32_to_f16_kernel<<<(NW + 255) / 256, 256, 0, stream>>>(wk, wkh, NW);
    f32_to_f16_kernel<<<(NW + 255) / 256, 256, 0, stream>>>(wv, wvh, NW);
    f32_to_f16_kernel<<<(NW + 255) / 256, 256, 0, stream>>>(wo, woh, NW);

    const int M = BATCH * SEQ;            // 8192
    dim3 gg(M / 128, EMBED / 64);         // (64, 16)
    gemm_wmma_kernel<<<gg, 256, 0, stream>>>(xh, wqh, bq, Qh, nullptr, M, EMBED, EMBED, 0);
    gemm_wmma_kernel<<<gg, 256, 0, stream>>>(xh, wkh, bk, Kh, nullptr, M, EMBED, EMBED, 0);
    gemm_wmma_kernel<<<gg, 256, 0, stream>>>(xh, wvh, bv, Vt, nullptr, M, EMBED, EMBED, 1);

    attn_wmma_kernel<<<dim3(SEQ / 128, BATCH * HEADS), 256, 0, stream>>>(Qh, Kh, Vt, Ah);

    gemm_wmma_kernel<<<gg, 256, 0, stream>>>(Ah, woh, bo, nullptr, out, M, EMBED, EMBED, 2);
}
